// RetrievalandRanking_71562745086099
// MI455X (gfx1250) — matrix operand order repaired and checked by reading, not compiled, not run on hardware
//
#include <hip/hip_runtime.h>

// Problem constants (from reference: Q=512, N=262144, D=768, k=10)
#define Q_TOTAL 512
#define N_TOTAL 262144
#define DIM     768
#define TOPK    10

#define QT      64                   // queries per block (grid.y = 8 -> lib L2 traffic 6.4 GB)
#define NT      32                   // lib rows per inner tile (staged in LDS as bf16)
#define NSUPER  64                   // n superblocks (grid.x)
#define NCHUNK  (N_TOTAL / NSUPER)   // 4096 lib rows per block
#define NITER   (NCHUNK / NT)        // 128 inner iterations
#define KSTEPS  (DIM / 32)           // 24 WMMA K-steps
#define LSTRIDE 776                  // 768 + 8 halfs pad -> conflict-free ds_load_b128
#define C4PERROW (DIM / 4)           // 192 float4 per row

typedef __bf16 v16bf __attribute__((ext_vector_type(16)));
typedef __bf16 v8bf  __attribute__((ext_vector_type(8)));
typedef __bf16 v4bf  __attribute__((ext_vector_type(4)));
typedef float  v8f   __attribute__((ext_vector_type(8)));

// ---------------------------------------------------------------------------
// Kernel 1: per-row sum of squares (one wave32 per row, coalesced float4)
// ---------------------------------------------------------------------------
__global__ __launch_bounds__(256)
void sumsq_kernel(const float* __restrict__ x, float* __restrict__ out, int nrows) {
    const int wave = threadIdx.x >> 5;
    const int lane = threadIdx.x & 31;
    const int row  = blockIdx.x * 8 + wave;
    if (row >= nrows) return;
    const float4* p = (const float4*)(x + (size_t)row * DIM);
    float s = 0.0f;
#pragma unroll
    for (int j = 0; j < DIM / 4 / 32; ++j) {  // 6 float4 per lane
        float4 v = p[lane + j * 32];
        s += v.x * v.x + v.y * v.y + v.z * v.z + v.w * v.w;
    }
#pragma unroll
    for (int off = 16; off > 0; off >>= 1) s += __shfl_xor(s, off, 32);
    if (lane == 0) out[row] = s;
}

// ---------------------------------------------------------------------------
// Kernel 2: fused bf16-WMMA distance GEMM + per-block top-10
//   block = 256 threads = 8 waves; tile = 64 queries x 32 lib rows per iter
//   wave: qsub = wave&3 (16 queries), nsub = wave>>2 (16 lib rows)
//   Both operands staged in LDS as bf16 (queries once, lib per iteration).
// ---------------------------------------------------------------------------
__global__ __launch_bounds__(256)
void disttopk_kernel(const float* __restrict__ query, const float* __restrict__ lib,
                     const float* __restrict__ qsq,   const float* __restrict__ xsq,
                     float* __restrict__ candd, int* __restrict__ candi) {
    __shared__ __align__(16) __bf16 qlds[QT * LSTRIDE];   // 97.0 KB (queries, full K)
    __shared__ __align__(16) __bf16 blds[NT * LSTRIDE];   // 48.5 KB (lib chunk, full K)
    __shared__ float d2s[QT][NT + 1];                     // 8.3 KB, padded
    __shared__ float topd[QT][TOPK];
    __shared__ int   topi[QT][TOPK];
    __shared__ float qsql[QT];

    const int tid  = threadIdx.x;
    const int wave = tid >> 5;
    const int lane = tid & 31;
    const int qsub = wave & 3;   // 0..3 -> 64 queries
    const int nsub = wave >> 2;  // 0..1 -> 32 lib rows
    const int qbase  = blockIdx.y * QT;
    const int nbase0 = blockIdx.x * NCHUNK;

    // Stage query tile fp32 -> bf16 into LDS (once per block), vectorized
#pragma unroll 4
    for (int t = 0; t < QT * C4PERROW / 256; ++t) {       // 48 float4 per thread
        int idx4 = tid + t * 256;
        int r    = idx4 / C4PERROW;
        int c4   = idx4 - r * C4PERROW;
        float4 f = ((const float4*)(query + (size_t)(qbase + r) * DIM))[c4];
        v4bf o = { (__bf16)f.x, (__bf16)f.y, (__bf16)f.z, (__bf16)f.w };
        *(v4bf*)(qlds + r * LSTRIDE + c4 * 4) = o;
    }
    if (tid < QT) {
        qsql[tid] = qsq[qbase + tid];
#pragma unroll
        for (int j = 0; j < TOPK; ++j) { topd[tid][j] = 3.0e38f; topi[tid][j] = -1; }
    }
    __syncthreads();

    const int halfsel = (lane < 16) ? 0 : 1;  // lane group selects K-half
    const int mrow    = lane & 15;

    for (int it = 0; it < NITER; ++it) {
        const int nbase = nbase0 + it * NT;

        // Stage lib chunk fp32 -> bf16 into LDS (once per block per iter)
#pragma unroll 4
        for (int t = 0; t < NT * C4PERROW / 256; ++t) {   // 24 float4 per thread
            int idx4 = tid + t * 256;
            int r    = idx4 / C4PERROW;
            int c4   = idx4 - r * C4PERROW;
            float4 f = ((const float4*)(lib + (size_t)(nbase + r) * DIM))[c4];
            v4bf o = { (__bf16)f.x, (__bf16)f.y, (__bf16)f.z, (__bf16)f.w };
            *(v4bf*)(blds + r * LSTRIDE + c4 * 4) = o;
        }
        // Prefetch next chunk while this one computes
        if (it + 1 < NITER)
            __builtin_prefetch(lib + (size_t)(nbase + NT + (tid >> 3)) * DIM + (tid & 7) * 96, 0, 1);
        __syncthreads();

        v8f acc = {};
#pragma unroll
        for (int ks = 0; ks < KSTEPS; ++ks) {
            const int kk = ks * 32;
            // A operand (16x32 bf16): lane<16 -> K 0-7 & 16-23; lane>=16 -> K 8-15 & 24-31
            const v8bf* arow =
                (const v8bf*)(qlds + (qsub * 16 + mrow) * LSTRIDE + kk + halfsel * 8);
            v8bf alo = arow[0];
            v8bf ahi = arow[2];  // +16 halfs
            v16bf a = __builtin_shufflevector(alo, ahi,
                0,1,2,3,4,5,6,7,8,9,10,11,12,13,14,15);
            // B operand (32x16 bf16): lane<16 -> K 0-7 & 16-23; lane>=16 -> K 8-15 & 24-31
            const v8bf* brow =
                (const v8bf*)(blds + (nsub * 16 + (lane & 15)) * LSTRIDE + kk + halfsel * 8);
            v8bf blo = brow[0];
            v8bf bhi = brow[2];  // +16 halfs
            v16bf b = __builtin_shufflevector(blo, bhi,
                0,1,2,3,4,5,6,7,8,9,10,11,12,13,14,15);

            acc = __builtin_amdgcn_wmma_f32_16x16x32_bf16(
                false, a, false, b, (short)0, acc, false, false);
        }

        // Epilogue: d2 = |q|^2 - 2*dot + |x|^2 ; C layout: M = r + 8*(lane>=16), N = lane%16
        const float xs = xsq[nbase + nsub * 16 + (lane & 15)];
#pragma unroll
        for (int r = 0; r < 8; ++r) {
            const int ml = qsub * 16 + r + halfsel * 8;
            d2s[ml][nsub * 16 + (lane & 15)] = qsql[ml] - 2.0f * acc[r] + xs;
        }
        __syncthreads();

        // Per-query sorted top-10 update (thread t owns query t -> race-free)
        if (tid < QT) {
            for (int j = 0; j < NT; ++j) {
                float v = d2s[tid][j];
                if (v < topd[tid][TOPK - 1]) {
                    int pos = TOPK - 1;
                    while (pos > 0 && topd[tid][pos - 1] > v) {
                        topd[tid][pos] = topd[tid][pos - 1];
                        topi[tid][pos] = topi[tid][pos - 1];
                        --pos;
                    }
                    topd[tid][pos] = v;
                    topi[tid][pos] = nbase + j;
                }
            }
        }
        __syncthreads();
    }

    // Emit this block's per-query top-10 candidates
    if (tid < QT) {
        const int q = qbase + tid;
        const size_t base = ((size_t)q * NSUPER + blockIdx.x) * TOPK;
#pragma unroll
        for (int j = 0; j < TOPK; ++j) {
            candd[base + j] = topd[tid][j];
            candi[base + j] = topi[tid][j];
        }
    }
}

// ---------------------------------------------------------------------------
// Kernel 3: reduce 64 partial top-10s per query -> final top-10
// ---------------------------------------------------------------------------
__global__ __launch_bounds__(32)
void topk_reduce_kernel(const float* __restrict__ candd, const int* __restrict__ candi,
                        float* __restrict__ dis, int* __restrict__ idc) {
    __shared__ float ld[32 * TOPK];
    __shared__ int   li[32 * TOPK];
    const int q    = blockIdx.x;
    const int lane = threadIdx.x;

    float bd[TOPK]; int bi[TOPK];
#pragma unroll
    for (int j = 0; j < TOPK; ++j) { bd[j] = 3.0e38f; bi[j] = -1; }

    const int total = NSUPER * TOPK;  // 640 candidates per query
    for (int c = lane; c < total; c += 32) {
        float cv = candd[(size_t)q * total + c];
        int   ci = candi[(size_t)q * total + c];
        if (cv < bd[TOPK - 1]) {
#pragma unroll
            for (int j = 0; j < TOPK; ++j) {  // register bubble-insert (stays in VGPRs)
                if (cv < bd[j]) {
                    float td = bd[j]; int ti = bi[j];
                    bd[j] = cv; bi[j] = ci;
                    cv = td; ci = ti;
                }
            }
        }
    }
#pragma unroll
    for (int j = 0; j < TOPK; ++j) { ld[lane * TOPK + j] = bd[j]; li[lane * TOPK + j] = bi[j]; }
    __syncthreads();

    if (lane == 0) {
        float fd[TOPK]; int fi[TOPK];
#pragma unroll
        for (int j = 0; j < TOPK; ++j) { fd[j] = ld[j]; fi[j] = li[j]; }
        for (int t = 1; t < 32; ++t) {
            for (int j = 0; j < TOPK; ++j) {
                float cv = ld[t * TOPK + j];
                int   ci = li[t * TOPK + j];
                if (cv >= fd[TOPK - 1]) break;  // lists are sorted ascending
#pragma unroll
                for (int s = 0; s < TOPK; ++s) {
                    if (cv < fd[s]) {
                        float td = fd[s]; int ti = fi[s];
                        fd[s] = cv; fi[s] = ci;
                        cv = td; ci = ti;
                    }
                }
            }
        }
#pragma unroll
        for (int j = 0; j < TOPK; ++j) {
            dis[q * TOPK + j] = fd[j];
            idc[q * TOPK + j] = fi[j];
        }
    }
}

// ---------------------------------------------------------------------------
extern "C" void kernel_launch(void* const* d_in, const int* in_sizes, int n_in,
                              void* d_out, int out_size, void* d_ws, size_t ws_size,
                              hipStream_t stream) {
    (void)in_sizes; (void)n_in; (void)out_size; (void)ws_size;
    const float* query = (const float*)d_in[0];
    const float* lib   = (const float*)d_in[1];
    // d_in[2] = top_k scalar (10), compile-time constant here.

    // Workspace layout (floats): xsq[N] | qsq[Q] | candd[Q*64*10] | candi[Q*64*10]
    float* ws    = (float*)d_ws;
    float* xsq   = ws;
    float* qsq   = ws + N_TOTAL;
    float* candd = ws + N_TOTAL + Q_TOTAL;
    int*   candi = (int*)(candd + (size_t)Q_TOTAL * NSUPER * TOPK);

    float* dis = (float*)d_out;                        // [512, 10] fp32
    int*   idc = (int*)d_out + Q_TOTAL * TOPK;         // [512, 10] int32 (concatenated)

    sumsq_kernel<<<N_TOTAL / 8, 256, 0, stream>>>(lib, xsq, N_TOTAL);
    sumsq_kernel<<<Q_TOTAL / 8, 256, 0, stream>>>(query, qsq, Q_TOTAL);

    dim3 grid(NSUPER, Q_TOTAL / QT);
    disttopk_kernel<<<grid, 256, 0, stream>>>(query, lib, qsq, xsq, candd, candi);

    topk_reduce_kernel<<<Q_TOTAL, 32, 0, stream>>>(candd, candi, dis, idc);
}
